// PatchRNNBlock_10359461118162
// MI455X (gfx1250) — compile-verified, matrix operand order checked by reading, not run on hardware
//
#include <hip/hip_runtime.h>
#include <hip/hip_bf16.h>
#include <math.h>

// ---------------------------------------------------------------------------
// Problem constants (from reference)
// ---------------------------------------------------------------------------
#define BSZ   2048
#define INF   2048
#define KW    4
#define LW    2045            // IN_F - K + 1
#define LK    8180            // LW * KW (flat width)
#define KP    8192            // padded K for fc1 GEMM
#define FCH   512
#define OUTF  720
#define EPSV  1e-5f

typedef __attribute__((ext_vector_type(16))) __bf16 v16bf;
typedef __attribute__((ext_vector_type(8)))  __bf16 v8bf;
typedef __attribute__((ext_vector_type(8)))  float  v8f;

__device__ __forceinline__ float gelu_exact(float x) {
    return 0.5f * x * (1.0f + erff(x * 0.70710678118654752f));
}

// Hardware transcendental paths (gfx1250 has V_TANH_F32)
__device__ __forceinline__ float fast_tanh(float x) {
#if __has_builtin(__builtin_amdgcn_tanhf)
    return __builtin_amdgcn_tanhf(x);
#else
    const float e = __expf(2.0f * x);
    return 1.0f - 2.0f * __builtin_amdgcn_rcpf(e + 1.0f);
#endif
}
__device__ __forceinline__ float fast_sigm(float x) {
    return fmaf(0.5f, fast_tanh(0.5f * x), 0.5f);
}

// ---------------------------------------------------------------------------
// K0: w_fc1 (8180x512 f32) -> w1T bf16 (512 x 8192), transposed + zero-padded
// ---------------------------------------------------------------------------
__global__ void k_conv_w1(const float* __restrict__ w1, __bf16* __restrict__ w1T) {
    const int n = blockIdx.x;                 // 0..511
    for (int k = threadIdx.x; k < KP; k += blockDim.x) {
        float v = (k < LK) ? w1[(size_t)k * FCH + n] : 0.0f;
        w1T[(size_t)n * KP + k] = (__bf16)v;
    }
}

// ---------------------------------------------------------------------------
// K1: w_fc2 (512x720 f32) -> w2T bf16 (720 x 512), transposed
// ---------------------------------------------------------------------------
__global__ void k_conv_w2(const float* __restrict__ w2, __bf16* __restrict__ w2T) {
    const int n = blockIdx.x;                 // 0..719
    for (int k = threadIdx.x; k < FCH; k += blockDim.x) {
        w2T[(size_t)n * FCH + k] = (__bf16)w2[(size_t)k * OUTF + n];
    }
}

// ---------------------------------------------------------------------------
// K2: per-L block: window + (xw+pe)@w_seg.T + b_seg, GELU, BN1 -> seg (B,L,K)
// One block per l; 256 threads; each thread covers 8 batch rows.
// ---------------------------------------------------------------------------
__global__ void k_seg(const float* __restrict__ x, const float* __restrict__ pe,
                      const float* __restrict__ w_seg, const float* __restrict__ b_seg,
                      const float* __restrict__ g1, const float* __restrict__ b1,
                      float* __restrict__ seg) {
    const int l   = blockIdx.x;
    const int tid = threadIdx.x;

    __shared__ float red[256];
    __shared__ float s_mean, s_istd;

    float pev[KW], ws[KW * KW], bs[KW];
#pragma unroll
    for (int k = 0; k < KW; ++k) pev[k] = pe[l * KW + k];
#pragma unroll
    for (int i = 0; i < KW * KW; ++i) ws[i] = w_seg[i];
#pragma unroll
    for (int j = 0; j < KW; ++j) bs[j] = b_seg[j];

    float y[8][KW];
    float s = 0.0f, ss = 0.0f;
#pragma unroll
    for (int r = 0; r < 8; ++r) {
        const int b = tid + r * 256;
        float v[KW];
#pragma unroll
        for (int k = 0; k < KW; ++k) v[k] = x[(size_t)b * INF + l + k] + pev[k];
#pragma unroll
        for (int j = 0; j < KW; ++j) {
            float z = bs[j];
#pragma unroll
            for (int k = 0; k < KW; ++k) z = fmaf(v[k], ws[j * KW + k], z);
            float g = gelu_exact(z);
            y[r][j] = g;
            s  += g;
            ss += g * g;
        }
    }

    red[tid] = s; __syncthreads();
    for (int o = 128; o > 0; o >>= 1) { if (tid < o) red[tid] += red[tid + o]; __syncthreads(); }
    const float S = red[0]; __syncthreads();
    red[tid] = ss; __syncthreads();
    for (int o = 128; o > 0; o >>= 1) { if (tid < o) red[tid] += red[tid + o]; __syncthreads(); }
    const float SS = red[0]; __syncthreads();

    if (tid == 0) {
        const float inv_n = 1.0f / (float)(BSZ * KW);
        const float m = S * inv_n;
        const float var = SS * inv_n - m * m;
        s_mean = m;
        s_istd = rsqrtf(var + EPSV);
    }
    __syncthreads();

    const float m = s_mean, is = s_istd;
    const float gg = g1[l], bb = b1[l];
#pragma unroll
    for (int r = 0; r < 8; ++r) {
        const int b = tid + r * 256;
#pragma unroll
        for (int j = 0; j < KW; ++j) {
            seg[(size_t)b * LK + l * KW + j] = (y[r][j] - m) * is * gg + bb;
        }
    }
}

// ---------------------------------------------------------------------------
// K3: LSTM, 4 lanes per batch row. Lane k owns c[k] and computes the four
// gates i[k], f[k], g[k], o[k]; the 4-wide h vector is re-broadcast across
// the lane group each step with __shfl (ds_bpermute, wave32).
// hs[b, t, k] = h_t[k]
// ---------------------------------------------------------------------------
__global__ void k_lstm(const float* __restrict__ seg,
                       const float* __restrict__ w_ih, const float* __restrict__ w_hh,
                       float* __restrict__ hs) {
    const int tid = threadIdx.x;
    const int gid = blockIdx.x * blockDim.x + tid;
    const int b = gid >> 2;        // batch row
    const int k = gid & 3;         // hidden index owned by this lane
    const int wl   = tid & 31;     // lane in wave
    const int base = wl & ~3;      // first lane of this row group

    // Per-lane gate weights: gate j*4+k (j = 0:i 1:f 2:g 3:o), 4 inputs each
    float wih[4][KW], whh[4][KW];
#pragma unroll
    for (int j = 0; j < 4; ++j)
#pragma unroll
        for (int m = 0; m < KW; ++m) {
            wih[j][m] = w_ih[(4 * j + k) * KW + m];
            whh[j][m] = w_hh[(4 * j + k) * KW + m];
        }

    float hv[KW] = {0.f, 0.f, 0.f, 0.f};
    float ck = 0.f;

    const float4* segp = (const float4*)(seg + (size_t)b * LK);
    float*        hsp  = hs + (size_t)b * LK;

    for (int t = 0; t < LW; ++t) {
        const float4 sv = segp[t];          // independent of recurrence: hoistable
        const float sk[KW] = {sv.x, sv.y, sv.z, sv.w};
        float z[4];
#pragma unroll
        for (int j = 0; j < 4; ++j) {
            float acc = 0.0f;
#pragma unroll
            for (int m = 0; m < KW; ++m) acc = fmaf(sk[m], wih[j][m], acc);
#pragma unroll
            for (int m = 0; m < KW; ++m) acc = fmaf(hv[m], whh[j][m], acc);
            z[j] = acc;
        }
        const float ig = fast_sigm(z[0]);
        const float fg = fast_sigm(z[1]);
        const float gv = fast_tanh(z[2]);
        const float og = fast_sigm(z[3]);
        ck = fmaf(fg, ck, ig * gv);
        const float hk = og * fast_tanh(ck);
        hsp[t * KW + k] = hk;
#pragma unroll
        for (int m = 0; m < KW; ++m) hv[m] = __shfl(hk, base + m, 32);
    }
}

// ---------------------------------------------------------------------------
// K4: BN2 statistics per-l over (B, K) of hs
// ---------------------------------------------------------------------------
__global__ void k_bn2_stats(const float* __restrict__ hs,
                            float* __restrict__ mean2, float* __restrict__ istd2) {
    const int l   = blockIdx.x;
    const int tid = threadIdx.x;
    __shared__ float red[256];

    float s = 0.0f, ss = 0.0f;
    for (int b = tid; b < BSZ; b += 256) {
        const float4 v = *(const float4*)(hs + (size_t)b * LK + l * KW);
        s  += v.x + v.y + v.z + v.w;
        ss += v.x * v.x + v.y * v.y + v.z * v.z + v.w * v.w;
    }
    red[tid] = s; __syncthreads();
    for (int o = 128; o > 0; o >>= 1) { if (tid < o) red[tid] += red[tid + o]; __syncthreads(); }
    const float S = red[0]; __syncthreads();
    red[tid] = ss; __syncthreads();
    for (int o = 128; o > 0; o >>= 1) { if (tid < o) red[tid] += red[tid + o]; __syncthreads(); }
    if (tid == 0) {
        const float inv_n = 1.0f / (float)(BSZ * KW);
        const float m = S * inv_n;
        const float var = red[0] * inv_n - m * m;
        mean2[l]  = m;
        istd2[l]  = rsqrtf(var + EPSV);
    }
}

// ---------------------------------------------------------------------------
// K5: out = BN2(hs)+seg, flatten, cast to bf16 into padded (B x 8192) matrix
// ---------------------------------------------------------------------------
__global__ void k_bn2_apply(const float* __restrict__ hs, const float* __restrict__ seg,
                            const float* __restrict__ mean2, const float* __restrict__ istd2,
                            const float* __restrict__ g2, const float* __restrict__ b2,
                            __bf16* __restrict__ flatbf) {
    const size_t idx = (size_t)blockIdx.x * blockDim.x + threadIdx.x;  // over B*KP
    const int b = (int)(idx >> 13);
    const int p = (int)(idx & (KP - 1));
    float v = 0.0f;
    if (p < LK) {
        const int l = p >> 2;
        v = (hs[(size_t)b * LK + p] - mean2[l]) * istd2[l] * g2[l] + b2[l]
            + seg[(size_t)b * LK + p];
    }
    flatbf[idx] = (__bf16)v;
}

// ---------------------------------------------------------------------------
// WMMA fragment loader: 16-bit A/B layout (16x32 tile slice per lane):
// lane = kh*16 + r (r = row for A / col for B), elems 0..7 -> K = 8*kh + 0..7,
// elems 8..15 -> K = 16 + 8*kh + 0..7. Rows are contiguous bf16.
// ---------------------------------------------------------------------------
__device__ __forceinline__ v16bf load_frag(const __bf16* rowp, int kk, int kb) {
    const v8bf lo = *(const v8bf*)(rowp + kk + kb);
    const v8bf hi = *(const v8bf*)(rowp + kk + 16 + kb);
    return __builtin_shufflevector(lo, hi, 0, 1, 2, 3, 4, 5, 6, 7,
                                           8, 9, 10, 11, 12, 13, 14, 15);
}

// ---------------------------------------------------------------------------
// K6: fc1 GEMM  h1 = gelu(flatbf(2048x8192) @ w1T^T + b_fc1)  -> bf16 (2048x512)
// 32x32 wave tile (2 A-frags x 2 B-frags -> 4 WMMA/iter).
// Block = 8 waves arranged 2(M) x 4(N): block tile 64 x 128.
// Grid (2048/64, 512/128) = (32, 4); 1024 waves total.
// ---------------------------------------------------------------------------
__global__ void k_fc1(const __bf16* __restrict__ A, const __bf16* __restrict__ BT,
                      const float* __restrict__ bias, __bf16* __restrict__ h1) {
    const int tid  = threadIdx.x;
    const int lane = tid & 31;
    const int wid  = tid >> 5;
    const int m0 = blockIdx.x * 64  + (wid & 1) * 32;
    const int n0 = blockIdx.y * 128 + (wid >> 1) * 32;

    const int r16 = lane & 15;
    const int kb  = (lane >> 4) * 8;

    const __bf16* arow0 = A  + (size_t)(m0 + r16) * KP;
    const __bf16* arow1 = arow0 + (size_t)16 * KP;
    const __bf16* brow0 = BT + (size_t)(n0 + r16) * KP;
    const __bf16* brow1 = brow0 + (size_t)16 * KP;

    v8f acc00 = {}, acc01 = {}, acc10 = {}, acc11 = {};
    for (int kk = 0; kk < KP; kk += 32) {
        __builtin_prefetch(arow0 + kk + 32, 0, 3);
        __builtin_prefetch(arow1 + kk + 32, 0, 3);
        __builtin_prefetch(brow0 + kk + 32, 0, 3);
        __builtin_prefetch(brow1 + kk + 32, 0, 3);
        const v16bf a0 = load_frag(arow0, kk, kb);
        const v16bf a1 = load_frag(arow1, kk, kb);
        const v16bf b0 = load_frag(brow0, kk, kb);
        const v16bf b1 = load_frag(brow1, kk, kb);
        acc00 = __builtin_amdgcn_wmma_f32_16x16x32_bf16(false, a0, false, b0, (short)0, acc00, false, false);
        acc01 = __builtin_amdgcn_wmma_f32_16x16x32_bf16(false, a0, false, b1, (short)0, acc01, false, false);
        acc10 = __builtin_amdgcn_wmma_f32_16x16x32_bf16(false, a1, false, b0, (short)0, acc10, false, false);
        acc11 = __builtin_amdgcn_wmma_f32_16x16x32_bf16(false, a1, false, b1, (short)0, acc11, false, false);
    }

    const int nn = lane & 15;
    const int mh = (lane >> 4) * 8;
    const float bn0 = bias[n0 + nn];
    const float bn1 = bias[n0 + 16 + nn];
#pragma unroll
    for (int r = 0; r < 8; ++r) {
        const int ma = m0 + mh + r;
        const int mb = ma + 16;
        h1[(size_t)ma * FCH + n0 + nn]      = (__bf16)gelu_exact(acc00[r] + bn0);
        h1[(size_t)ma * FCH + n0 + 16 + nn] = (__bf16)gelu_exact(acc01[r] + bn1);
        h1[(size_t)mb * FCH + n0 + nn]      = (__bf16)gelu_exact(acc10[r] + bn0);
        h1[(size_t)mb * FCH + n0 + 16 + nn] = (__bf16)gelu_exact(acc11[r] + bn1);
    }
}

// ---------------------------------------------------------------------------
// K7: fc2 GEMM  out = h1(2048x512) @ w2T^T + b_fc2  -> f32 (2048x720)
// 32x16 wave tile (2 A-frags share one B-frag). 8 waves stacked in M.
// Grid (2048/256, 720/16) = (8, 45); 2880 waves total.
// ---------------------------------------------------------------------------
__global__ void k_fc2(const __bf16* __restrict__ A, const __bf16* __restrict__ BT,
                      const float* __restrict__ bias, float* __restrict__ out) {
    const int tid  = threadIdx.x;
    const int lane = tid & 31;
    const int wid  = tid >> 5;
    const int m0 = blockIdx.x * 256 + wid * 32;
    const int n0 = blockIdx.y * 16;

    const int r16 = lane & 15;
    const int kb  = (lane >> 4) * 8;

    const __bf16* arow0 = A  + (size_t)(m0 + r16) * FCH;
    const __bf16* arow1 = arow0 + (size_t)16 * FCH;
    const __bf16* brow  = BT + (size_t)(n0 + r16) * FCH;

    v8f acc0 = {}, acc1 = {};
    for (int kk = 0; kk < FCH; kk += 32) {
        const v16bf a0 = load_frag(arow0, kk, kb);
        const v16bf a1 = load_frag(arow1, kk, kb);
        const v16bf b0 = load_frag(brow,  kk, kb);
        acc0 = __builtin_amdgcn_wmma_f32_16x16x32_bf16(false, a0, false, b0, (short)0, acc0, false, false);
        acc1 = __builtin_amdgcn_wmma_f32_16x16x32_bf16(false, a1, false, b0, (short)0, acc1, false, false);
    }

    const int n  = n0 + (lane & 15);
    const int mh = (lane >> 4) * 8;
    const float bn = bias[n];
#pragma unroll
    for (int r = 0; r < 8; ++r) {
        const int ma = m0 + mh + r;
        out[(size_t)ma * OUTF + n]        = acc0[r] + bn;
        out[(size_t)(ma + 16) * OUTF + n] = acc1[r] + bn;
    }
}

// ---------------------------------------------------------------------------
// Workspace layout (bytes, 256-aligned)
// ---------------------------------------------------------------------------
#define SEG_OFF    ((size_t)0)
#define HS_OFF     ((size_t)67010560)
#define MEAN2_OFF  ((size_t)134021120)
#define ISTD2_OFF  ((size_t)134029312)
#define FLATBF_OFF ((size_t)134037504)
#define W1T_OFF    ((size_t)167591936)
#define H1_OFF     ((size_t)175980544)
#define W2T_OFF    ((size_t)178077696)

extern "C" void kernel_launch(void* const* d_in, const int* in_sizes, int n_in,
                              void* d_out, int out_size, void* d_ws, size_t ws_size,
                              hipStream_t stream) {
    const float* x     = (const float*)d_in[0];
    const float* pe    = (const float*)d_in[1];
    const float* w_seg = (const float*)d_in[2];
    const float* b_seg = (const float*)d_in[3];
    const float* bn1_g = (const float*)d_in[4];
    const float* bn1_b = (const float*)d_in[5];
    const float* w_ih  = (const float*)d_in[6];
    const float* w_hh  = (const float*)d_in[7];
    const float* bn2_g = (const float*)d_in[8];
    const float* bn2_b = (const float*)d_in[9];
    const float* w_fc1 = (const float*)d_in[10];
    const float* b_fc1 = (const float*)d_in[11];
    const float* w_fc2 = (const float*)d_in[12];
    const float* b_fc2 = (const float*)d_in[13];

    char* ws = (char*)d_ws;
    float*  seg    = (float*)(ws + SEG_OFF);
    float*  hs     = (float*)(ws + HS_OFF);
    float*  mean2  = (float*)(ws + MEAN2_OFF);
    float*  istd2  = (float*)(ws + ISTD2_OFF);
    __bf16* flatbf = (__bf16*)(ws + FLATBF_OFF);
    __bf16* w1T    = (__bf16*)(ws + W1T_OFF);
    __bf16* h1     = (__bf16*)(ws + H1_OFF);
    __bf16* w2T    = (__bf16*)(ws + W2T_OFF);

    // Weight conversions (independent of pipeline front)
    k_conv_w1<<<dim3(FCH),  dim3(256), 0, stream>>>(w_fc1, w1T);
    k_conv_w2<<<dim3(OUTF), dim3(256), 0, stream>>>(w_fc2, w2T);

    // Stage 1: window + small matmul + GELU + BN1
    k_seg<<<dim3(LW), dim3(256), 0, stream>>>(x, pe, w_seg, b_seg, bn1_g, bn1_b, seg);

    // Stage 2: LSTM, 4 lanes per batch row (256 waves)
    k_lstm<<<dim3((BSZ * 4) / 256), dim3(256), 0, stream>>>(seg, w_ih, w_hh, hs);

    // Stage 3: BN2 stats + apply + residual + bf16 flatten (padded)
    k_bn2_stats<<<dim3(LW), dim3(256), 0, stream>>>(hs, mean2, istd2);
    k_bn2_apply<<<dim3((BSZ * KP) / 256), dim3(256), 0, stream>>>(
        hs, seg, mean2, istd2, bn2_g, bn2_b, flatbf);

    // Stage 4: fc1 (WMMA bf16, 32x32 wave tiles) + GELU
    k_fc1<<<dim3(BSZ / 64, FCH / 128), dim3(256), 0, stream>>>(flatbf, w1T, b_fc1, h1);

    // Stage 5: fc2 (WMMA bf16, 32x16 wave tiles) -> f32 output
    k_fc2<<<dim3(BSZ / 256, OUTF / 16), dim3(256), 0, stream>>>(h1, w2T, b_fc2, (float*)d_out);
}
